// AttentionModule_19980187861835
// MI455X (gfx1250) — matrix operand order repaired and checked by reading, not compiled, not run on hardware
//
#include <hip/hip_runtime.h>
#include <hip/hip_bf16.h>

// ---------------------------------------------------------------------------
// MI455X (gfx1250, wave32) implementation.
//
// Identity: softmax(x, axis=n).sum(axis=n) == 1 (Z/Z summed, ~1e-6 in fp32),
// so s1 = s2 = 1, p1 = q, att = v.  The q/k GEMMs are dead code below the
// bf16 tensor-core noise floor and are elided.
//
// Pipeline:
//   prep:    fp32 input -> zero-padded [16][66][66][128] bf16 (halo=1),
//            transpose conv/v weights to [N][K] bf16
//   conv3x3: implicit-GEMM, v_wmma_f32_16x16x32_bf16, M=65536 N=256 K=1152,
//            double-buffered LDS fed by global_load_async_to_lds_b128
//   bnstats: per-channel sum/sumsq, deterministic 2-stage
//   bnapply: scale/shift + LeakyReLU(0.3) -> x fp32 (+ bf16 copy)
//   vgemm:   1x1 conv (256->256), same WMMA + async-LDS pipeline
//   y=x+v + per-sample LN stats (2-stage), LN apply + LeakyReLU -> out
// ---------------------------------------------------------------------------

typedef __attribute__((ext_vector_type(16))) __bf16 v16bf;
typedef __attribute__((ext_vector_type(8)))  __bf16 v8bf;
typedef __attribute__((ext_vector_type(8)))  float  v8f;

#define WMMA_BF16(a, b, c) \
  __builtin_amdgcn_wmma_f32_16x16x32_bf16(false, (a), false, (b), (short)0, (c), false, false)

// ---- CDNA5 async global->LDS staging (ASYNCcnt) ---------------------------
// Each stage issues exactly 3 async b128 per thread (1 for A, 2 for B).
// Async loads of one wave complete in order, so s_wait_asynccnt 3 while the
// next stage is in flight guarantees the current stage has landed in LDS.
#define USE_ASYNC_COPY 1

#if USE_ASYNC_COPY
#define ASYNC_COPY16(g, l)                                                   \
  asm volatile("global_load_async_to_lds_b128 %0, %1, off"                   \
               :: "v"((unsigned)(size_t)(l)),                                \
                  "v"((unsigned long long)(size_t)(g))                       \
               : "memory")
#define WAIT_ASYNC(n) asm volatile("s_wait_asynccnt " #n ::: "memory")
#else
#define ASYNC_COPY16(g, l) (*(uint4*)(l) = *(const uint4*)(g))
#define WAIT_ASYNC(n)
#endif

#define B_    16
#define HP_   66                 // padded height/width (halo = 1)
#define CIN_  128
#define C_    256
#define NROWS (B_ * 64 * 64)     // 65536
#define KCONV (9 * CIN_)         // 1152
#define NTOT  (NROWS * C_)       // 16777216
#define NPAD  (B_ * HP_ * HP_ * CIN_)  // 8,921,088

// ---------------- fragment loaders (ISA 7.12.2 layouts, wave32) -------------
// A 16x32 bf16: lane<16 -> M=lane, K 0..7 (v0-3) & 16..23 (v4-7);
//               lane>=16 -> M=lane-16, K 8..15 & 24..31.
__device__ inline v16bf load_a_frag(const __bf16* As /*[64][32]*/, int mBase) {
  const int lane = threadIdx.x & 31;
  const int half = lane >> 4, mr = lane & 15;
  const __bf16* rowp = As + (size_t)(mBase + mr) * 32;
  v8bf lo = *(const v8bf*)(rowp + half * 8);
  v8bf hi = *(const v8bf*)(rowp + 16 + half * 8);
  return __builtin_shufflevector(lo, hi, 0, 1, 2, 3, 4, 5, 6, 7,
                                 8, 9, 10, 11, 12, 13, 14, 15);
}
// B 32x16 bf16 stored transposed in LDS as [n][k]:
//   lane<16 -> N=lane, K 0..7 & 16..23; lane>=16 -> N=lane-16, K 8..15 & 24..31.
__device__ inline v16bf load_b_frag(const __bf16* Bs /*[128][32]*/, int nBase) {
  const int lane = threadIdx.x & 31;
  const int half = lane >> 4, nc = lane & 15;
  const __bf16* rowp = Bs + (size_t)(nBase + nc) * 32;
  v8bf lo = *(const v8bf*)(rowp + half * 8);
  v8bf hi = *(const v8bf*)(rowp + 16 + half * 8);
  return __builtin_shufflevector(lo, hi, 0, 1, 2, 3, 4, 5, 6, 7,
                                 8, 9, 10, 11, 12, 13, 14, 15);
}

// ---------------- prep kernels ---------------------------------------------
// fp32 [16][64][64][128] -> bf16 [16][66][66][128], zero halo.
__global__ void pad_cvt_kernel(const float* __restrict__ in,
                               __bf16* __restrict__ out) {
  for (int i = blockIdx.x * blockDim.x + threadIdx.x; i < NPAD;
       i += gridDim.x * blockDim.x) {
    const int c    = i & 127;
    const int rest = i >> 7;              // b*66*66 + h2*66 + w2
    const int w2 = rest % HP_;
    const int r2 = rest / HP_;
    const int h2 = r2 % HP_;
    const int b  = r2 / HP_;
    const int h = h2 - 1, w = w2 - 1;
    float v = 0.f;
    if ((unsigned)h < 64u && (unsigned)w < 64u)
      v = in[(size_t)(((b * 64 + h) * 64 + w) * 128 + c)];
    out[i] = (__bf16)v;
  }
}

// in[k*256 + n] (fp32) -> out[n*K + k] (bf16).
__global__ void transpose_w_kernel(const float* __restrict__ in,
                                   __bf16* __restrict__ out, int K) {
  for (int i = blockIdx.x * blockDim.x + threadIdx.x; i < K * 256;
       i += gridDim.x * blockDim.x) {
    int n = i & 255, k = i >> 8;
    out[(size_t)n * K + k] = (__bf16)in[i];
  }
}

// ---------------- conv 3x3 implicit GEMM (WMMA bf16, async pipeline) -------
__global__ void __launch_bounds__(256)
conv3x3_wmma_kernel(const __bf16* __restrict__ xpad,  // [16][66][66][128] bf16
                    const __bf16* __restrict__ wt,    // [C][KCONV] bf16
                    const float* __restrict__ bias,
                    float* __restrict__ out) {        // [NROWS][C]
  __shared__ alignas(64) __bf16 As[2][64 * 32];
  __shared__ alignas(64) __bf16 Bs[2][128 * 32];

  const int t     = threadIdx.x;
  const int wave  = t >> 5;
  const int wm    = wave >> 2;          // 0..1  (32 rows)
  const int wn    = wave & 3;           // 0..3  (32 cols)
  const int mBase = blockIdx.x * 64;
  const int nBase = blockIdx.y * 128;

  v8f acc[2][2] = {};

  // A staging: thread t -> row am, 16B chunk ak (branch-free: padded input).
  const int am  = t >> 2;
  const int ak  = (t & 3) * 8;
  const int gm  = mBase + am;
  const int b   = gm >> 12;
  const int rem = gm & 4095;
  const int h   = rem >> 6;
  const int w   = rem & 63;
  // B staging: thread t -> 2 chunks.
  const int bn0 = t >> 2,           bk0 = (t & 3) * 8;
  const int bn1 = (t + 256) >> 2,   bk1 = bk0;

  auto issue = [&](int it, int buf) {
    const int k0    = it * 32;
    const int plane = k0 >> 7;          // kh*3+kw; K_TILE=32 stays in-plane
    const int cinB  = k0 & 127;
    const int kh = plane / 3, kw = plane % 3;
    const __bf16* ga =
        xpad + (size_t)(((b * HP_ + h + kh) * HP_ + (w + kw)) * CIN_ + cinB + ak);
    ASYNC_COPY16(ga, &As[buf][am * 32 + ak]);
    ASYNC_COPY16(wt + (size_t)(nBase + bn0) * KCONV + k0 + bk0,
                 &Bs[buf][bn0 * 32 + bk0]);
    ASYNC_COPY16(wt + (size_t)(nBase + bn1) * KCONV + k0 + bk1,
                 &Bs[buf][bn1 * 32 + bk1]);
  };

  issue(0, 0);
  for (int it = 0; it < 36; ++it) {
    const int buf = it & 1;
    if (it + 1 < 36) { issue(it + 1, buf ^ 1); WAIT_ASYNC(3); }
    else             { WAIT_ASYNC(0); }
    __syncthreads();
    v16bf a0 = load_a_frag(As[buf], wm * 32);
    v16bf a1 = load_a_frag(As[buf], wm * 32 + 16);
    v16bf b0 = load_b_frag(Bs[buf], wn * 32);
    v16bf b1 = load_b_frag(Bs[buf], wn * 32 + 16);
    acc[0][0] = WMMA_BF16(a0, b0, acc[0][0]);
    acc[0][1] = WMMA_BF16(a0, b1, acc[0][1]);
    acc[1][0] = WMMA_BF16(a1, b0, acc[1][0]);
    acc[1][1] = WMMA_BF16(a1, b1, acc[1][1]);
    __syncthreads();
  }

  // Epilogue: + bias.  C layout: m = 8*(lane>>4)+r, n = lane&15.
  const int lane = t & 31;
  const int half = lane >> 4, nc = lane & 15;
  #pragma unroll
  for (int mi = 0; mi < 2; ++mi)
    #pragma unroll
    for (int ni = 0; ni < 2; ++ni) {
      const int n = nBase + wn * 32 + ni * 16 + nc;
      const float bv = bias[n];
      #pragma unroll
      for (int r = 0; r < 8; ++r) {
        const int m = mBase + wm * 32 + mi * 16 + half * 8 + r;
        out[(size_t)m * C_ + n] = acc[mi][ni][r] + bv;
      }
    }
}

// ---------------- 1x1 conv (v) GEMM, K=256 (WMMA bf16, async pipeline) -----
__global__ void __launch_bounds__(256)
gemm256_wmma_kernel(const __bf16* __restrict__ xbf,  // [NROWS][256] bf16
                    const __bf16* __restrict__ wt,   // [256][256] bf16 [n][k]
                    const float* __restrict__ bias,
                    float* __restrict__ out) {
  __shared__ alignas(64) __bf16 As[2][64 * 32];
  __shared__ alignas(64) __bf16 Bs[2][128 * 32];

  const int t     = threadIdx.x;
  const int wave  = t >> 5;
  const int wm    = wave >> 2;
  const int wn    = wave & 3;
  const int mBase = blockIdx.x * 64;
  const int nBase = blockIdx.y * 128;

  v8f acc[2][2] = {};
  const int am = t >> 2,          ak  = (t & 3) * 8;
  const int bn0 = t >> 2,         bk0 = ak;
  const int bn1 = (t + 256) >> 2, bk1 = ak;

  auto issue = [&](int it, int buf) {
    const int k0 = it * 32;
    ASYNC_COPY16(xbf + (size_t)(mBase + am) * 256 + k0 + ak,
                 &As[buf][am * 32 + ak]);
    ASYNC_COPY16(wt + (size_t)(nBase + bn0) * 256 + k0 + bk0,
                 &Bs[buf][bn0 * 32 + bk0]);
    ASYNC_COPY16(wt + (size_t)(nBase + bn1) * 256 + k0 + bk1,
                 &Bs[buf][bn1 * 32 + bk1]);
  };

  issue(0, 0);
  for (int it = 0; it < 8; ++it) {
    const int buf = it & 1;
    if (it + 1 < 8) { issue(it + 1, buf ^ 1); WAIT_ASYNC(3); }
    else            { WAIT_ASYNC(0); }
    __syncthreads();
    v16bf a0 = load_a_frag(As[buf], wm * 32);
    v16bf a1 = load_a_frag(As[buf], wm * 32 + 16);
    v16bf b0 = load_b_frag(Bs[buf], wn * 32);
    v16bf b1 = load_b_frag(Bs[buf], wn * 32 + 16);
    acc[0][0] = WMMA_BF16(a0, b0, acc[0][0]);
    acc[0][1] = WMMA_BF16(a0, b1, acc[0][1]);
    acc[1][0] = WMMA_BF16(a1, b0, acc[1][0]);
    acc[1][1] = WMMA_BF16(a1, b1, acc[1][1]);
    __syncthreads();
  }

  const int lane = t & 31;
  const int half = lane >> 4, nc = lane & 15;
  #pragma unroll
  for (int mi = 0; mi < 2; ++mi)
    #pragma unroll
    for (int ni = 0; ni < 2; ++ni) {
      const int n = nBase + wn * 32 + ni * 16 + nc;
      const float bv = bias[n];
      #pragma unroll
      for (int r = 0; r < 8; ++r) {
        const int m = mBase + wm * 32 + mi * 16 + half * 8 + r;
        out[(size_t)m * C_ + n] = acc[mi][ni][r] + bv;
      }
    }
}

// ---------------- BN stats: per-channel partials (deterministic) -----------
__global__ void bn_stats_kernel(const float* __restrict__ conv,
                                float* __restrict__ part,    // [512][256]
                                float* __restrict__ part2) {
  const int c = threadIdx.x;
  const int rowBase = blockIdx.x * 128;
  float s = 0.f, s2 = 0.f;
  for (int r = 0; r < 128; ++r) {
    float v = conv[(size_t)(rowBase + r) * C_ + c];
    s += v; s2 += v * v;
  }
  part[blockIdx.x * C_ + c]  = s;
  part2[blockIdx.x * C_ + c] = s2;
}

__global__ void bn_finalize_kernel(const float* __restrict__ part,
                                   const float* __restrict__ part2,
                                   const float* __restrict__ gamma,
                                   const float* __restrict__ beta,
                                   float* __restrict__ sclshift) { // [512]
  const int c = threadIdx.x;
  float s = 0.f, s2 = 0.f;
  for (int k = 0; k < 512; ++k) { s += part[k * C_ + c]; s2 += part2[k * C_ + c]; }
  const float inv = 1.0f / (float)NROWS;
  const float mu  = s * inv;
  const float var = s2 * inv - mu * mu;
  const float scl = gamma[c] * rsqrtf(var + 1e-3f);
  sclshift[c]      = scl;
  sclshift[C_ + c] = beta[c] - mu * scl;
}

__global__ void bn_apply_kernel(const float* __restrict__ conv,
                                const float* __restrict__ sclshift,
                                float* __restrict__ xf,
                                __bf16* __restrict__ xb) {
  for (int i = blockIdx.x * blockDim.x + threadIdx.x; i < NTOT;
       i += gridDim.x * blockDim.x) {
    const int c = i & 255;
    float v = conv[i] * sclshift[c] + sclshift[C_ + c];
    v = v > 0.f ? v : 0.3f * v;
    xf[i] = v;
    xb[i] = (__bf16)v;
  }
}

// ---------------- y = x + v, per-sample LN partials (deterministic) --------
__global__ void add_lnstats_kernel(const float* __restrict__ x,
                                   const float* __restrict__ v,
                                   float* __restrict__ y,       // may alias x
                                   float* __restrict__ lnpart,  // [16][64]
                                   float* __restrict__ lnpart2) {
  const int b = blockIdx.x;
  const size_t base = ((size_t)b << 20) + (size_t)blockIdx.y * 16384;
  float s = 0.f, s2 = 0.f;
  for (int i = threadIdx.x; i < 16384; i += 256) {
    float val = x[base + i] + v[base + i];
    y[base + i] = val;
    s += val; s2 += val * val;
  }
  __shared__ float rs[256], rs2[256];
  rs[threadIdx.x] = s; rs2[threadIdx.x] = s2;
  __syncthreads();
  for (int off = 128; off > 0; off >>= 1) {
    if (threadIdx.x < off) {
      rs[threadIdx.x]  += rs[threadIdx.x + off];
      rs2[threadIdx.x] += rs2[threadIdx.x + off];
    }
    __syncthreads();
  }
  if (threadIdx.x == 0) {
    lnpart[b * 64 + blockIdx.y]  = rs[0];
    lnpart2[b * 64 + blockIdx.y] = rs2[0];
  }
}

__global__ void ln_finalize_kernel(const float* __restrict__ lnpart,
                                   const float* __restrict__ lnpart2,
                                   float* __restrict__ lnstats) { // [32]
  const int b = threadIdx.x;   // 16 threads
  float s = 0.f, s2 = 0.f;
  for (int i = 0; i < 64; ++i) { s += lnpart[b * 64 + i]; s2 += lnpart2[b * 64 + i]; }
  lnstats[b]      = s;
  lnstats[16 + b] = s2;
}

__global__ void ln_apply_kernel(const float* __restrict__ y,
                                const float* __restrict__ lnstats,
                                const float* __restrict__ lng,
                                const float* __restrict__ lnb,
                                float* __restrict__ out) {
  const float inv = 1.0f / 1048576.0f;
  for (int i = blockIdx.x * blockDim.x + threadIdx.x; i < NTOT;
       i += gridDim.x * blockDim.x) {
    const int b   = i >> 20;
    const int hwc = i & 0xFFFFF;
    const float mu  = lnstats[b] * inv;
    const float var = lnstats[16 + b] * inv - mu * mu;
    const float r   = rsqrtf(var + 1e-3f);
    float val = (y[i] - mu) * r * lng[hwc] + lnb[hwc];
    out[i] = val > 0.f ? val : 0.3f * val;
  }
}

// ---------------------------------------------------------------------------
extern "C" void kernel_launch(void* const* d_in, const int* in_sizes, int n_in,
                              void* d_out, int out_size, void* d_ws, size_t ws_size,
                              hipStream_t stream) {
  (void)in_sizes; (void)n_in; (void)out_size; (void)ws_size;

  const float* inputs = (const float*)d_in[0];   // (16,64,64,128)
  const float* cbl_w  = (const float*)d_in[1];   // (3,3,128,256)
  const float* cbl_b  = (const float*)d_in[2];
  const float* bn_g   = (const float*)d_in[3];
  const float* bn_b   = (const float*)d_in[4];
  // d_in[5..8] = wq,bq,wk,bk — dead: softmax.sum over its own axis == 1.
  const float* wv     = (const float*)d_in[9];   // (1,1,256,256)
  const float* bv     = (const float*)d_in[10];
  const float* lng    = (const float*)d_in[11];  // (64,64,256)
  const float* lnb    = (const float*)d_in[12];

  char* ws = (char*)d_ws;
  __bf16* xpad    = (__bf16*)(ws);                       // 17,842,176 B
  __bf16* wconv_t = (__bf16*)(ws + 17842176);            //    589,824 B
  __bf16* wv_t    = (__bf16*)(ws + 18432000);            //    131,072 B
  float*  conv    = (float*) (ws + 18563072);            // 67,108,864 B (= x = y)
  __bf16* x_bf    = (__bf16*)(ws + 85671936);            // 33,554,432 B
  float*  vout    = (float*) (ws + 119226368);           // 67,108,864 B
  float*  stats   = (float*) (ws + 186335232);           //  ~1.06 MB
  float* bnpart   = stats;               // 131072
  float* bnpart2  = stats + 131072;      // 131072
  float* sclshift = stats + 262144;      // 512
  float* lnpart   = stats + 262656;      // 1024
  float* lnpart2  = stats + 263680;      // 1024
  float* lnstats  = stats + 264704;      // 32

  // prep
  pad_cvt_kernel<<<4096, 256, 0, stream>>>(inputs, xpad);
  transpose_w_kernel<<<1152, 256, 0, stream>>>(cbl_w, wconv_t, KCONV);
  transpose_w_kernel<<<256, 256, 0, stream>>>(wv, wv_t, 256);

  // conv3x3 (WMMA) -> conv fp32
  conv3x3_wmma_kernel<<<dim3(NROWS / 64, C_ / 128), 256, 0, stream>>>(
      xpad, wconv_t, cbl_b, conv);

  // batch-norm (train stats) + leaky -> x (fp32 in place) + bf16 copy
  bn_stats_kernel<<<512, 256, 0, stream>>>(conv, bnpart, bnpart2);
  bn_finalize_kernel<<<1, 256, 0, stream>>>(bnpart, bnpart2, bn_g, bn_b, sclshift);
  bn_apply_kernel<<<4096, 256, 0, stream>>>(conv, sclshift, conv, x_bf);

  // v = 1x1 conv (WMMA);  att = v since s2 == 1
  gemm256_wmma_kernel<<<dim3(NROWS / 64, C_ / 128), 256, 0, stream>>>(
      x_bf, wv_t, bv, vout);

  // y = x + v (in place over x), then LayerNorm + leaky -> out
  add_lnstats_kernel<<<dim3(B_, 64), 256, 0, stream>>>(conv, vout, conv,
                                                       lnpart, lnpart2);
  ln_finalize_kernel<<<1, 16, 0, stream>>>(lnpart, lnpart2, lnstats);
  ln_apply_kernel<<<4096, 256, 0, stream>>>(conv, lnstats, lng, lnb,
                                            (float*)d_out);
}
